// MultiHeadAttention_47244640256535
// MI455X (gfx1250) — matrix operand order repaired and checked by reading, not compiled, not run on hardware
//
#include <hip/hip_runtime.h>
#include <hip/hip_bf16.h>

// ---------------------------------------------------------------------------
// MultiHeadAttention (pre-LN self-attention, causal + key-padding mask)
// B=8, SQ=SKV=1024, D=1024, H=16, DQ=DV=64
// fp32 LN/softmax/epilogues; v_wmma_f32_16x16x32_f16 for all matmuls.
// GEMMs: register double-buffered (software pipelined), 32x64 tile per wave.
// Attention: flash, K/V chunks staged to LDS via async global->LDS copies
// (double buffered, one chunk ahead) when the builtin is available.
// ---------------------------------------------------------------------------

typedef __attribute__((ext_vector_type(16))) _Float16 v16h;
typedef __attribute__((ext_vector_type(8)))  _Float16 v8h;
typedef __attribute__((ext_vector_type(8)))  float    v8f;
typedef __attribute__((ext_vector_type(4)))  int      v4i;

typedef __attribute__((address_space(1))) v4i g_v4i;  // global 16B chunk
typedef __attribute__((address_space(3))) v4i l_v4i;  // LDS 16B chunk

union AB { v16h v; v8h h[2]; };

#define NB 8
#define NSQ 1024
#define ND 1024
#define NH 16
#define NDH 64
#define NROWS (NB * NSQ)   // 8192

#if defined(__has_builtin)
#if __has_builtin(__builtin_amdgcn_global_load_async_to_lds_b128) && \
    __has_builtin(__builtin_amdgcn_s_wait_asynccnt)
#define HAVE_ASYNC_LDS 1
#endif
#endif
#ifndef HAVE_ASYNC_LDS
#define HAVE_ASYNC_LDS 0
#endif

// copy 16 bytes global -> LDS (async if available)
__device__ __forceinline__ void async_cp16(void* lds_dst, const void* gsrc) {
#if HAVE_ASYNC_LDS
  __builtin_amdgcn_global_load_async_to_lds_b128((g_v4i*)gsrc, (l_v4i*)lds_dst,
                                                 0, 0);
#else
  *(v8h*)lds_dst = *(const v8h*)gsrc;
#endif
}
template <int N>
__device__ __forceinline__ void wait_async() {
#if HAVE_ASYNC_LDS
  __builtin_amdgcn_s_wait_asynccnt(N);
#endif
}

// A-operand (16x32 f16): lane<16 -> K {0..7,16..23}; lane>=16 -> K {8..15,24..31}
__device__ __forceinline__ v16h load_a16(const _Float16* base, int lh) {
  AB u;
  u.h[0] = *(const v8h*)(base + 8 * lh);
  u.h[1] = *(const v8h*)(base + 16 + 8 * lh);
  return u.v;
}
// B-operand (32x16 f16): lane<16 -> K {0..7,16..23}; lane>=16 -> K {8..15,24..31}
__device__ __forceinline__ v16h load_b16(const _Float16* base, int lh) {
  AB u;
  u.h[0] = *(const v8h*)(base + 8 * lh);
  u.h[1] = *(const v8h*)(base + 16 + 8 * lh);
  return u.v;
}

__device__ __forceinline__ v8f wmma_f16(v16h a, v16h b, v8f c) {
  return __builtin_amdgcn_wmma_f32_16x16x32_f16(false, a, false, b, (short)0, c,
                                                false, false);
}

// ---------------------------------------------------------------------------
// 1) LayerNorm: one wave per row of x[8192][1024]; writes fp32 + f16 copies.
// ---------------------------------------------------------------------------
__global__ void ln_kernel(const float* __restrict__ x,
                          const float* __restrict__ gamma,
                          const float* __restrict__ beta,
                          float* __restrict__ lnf,
                          _Float16* __restrict__ lnh) {
  int wave = threadIdx.x >> 5;
  int lane = threadIdx.x & 31;
  int row = blockIdx.x * 8 + wave;
  const float* xr = x + (size_t)row * ND;
  float s = 0.f, ss = 0.f;
  for (int i = lane; i < ND; i += 32) {
    float v = xr[i];
    s += v;
    ss += v * v;
  }
  for (int m = 16; m >= 1; m >>= 1) {
    s  += __shfl_xor(s, m, 32);
    ss += __shfl_xor(ss, m, 32);
  }
  float mean = s * (1.0f / ND);
  float var  = ss * (1.0f / ND) - mean * mean;
  float inv  = rsqrtf(var + 1e-3f);
  for (int i = lane; i < ND; i += 32) {
    float v = (xr[i] - mean) * inv * gamma[i] + beta[i];
    lnf[(size_t)row * ND + i] = v;
    lnh[(size_t)row * ND + i] = (_Float16)v;
  }
}

// ---------------------------------------------------------------------------
// 2) Weight convert + transpose: Wt[n][k] = (f16) W[k][n]
// ---------------------------------------------------------------------------
__global__ void wt_kernel(const float* __restrict__ W,
                          _Float16* __restrict__ Wt, int K, int N) {
  int idx = blockIdx.x * blockDim.x + threadIdx.x;
  if (idx >= K * N) return;
  int k = idx / N, n = idx - k * N;
  Wt[(size_t)n * K + k] = (_Float16)W[idx];
}

// ---------------------------------------------------------------------------
// Shared GEMM body: one wave -> 32x64 output tile (2x4 accumulators),
// register double-buffered so loads for k+32 overlap WMMAs for k.
// A: MxK f16 row-major, Bt: NxK f16 row-major (B transposed)
// ---------------------------------------------------------------------------
__device__ __forceinline__ void gemm32x64(const _Float16* __restrict__ A, int lda,
                                          const _Float16* __restrict__ Bt, int ldb,
                                          int K, int m0, int n0, int lane,
                                          v8f acc[2][4]) {
  int lh = lane >> 4, mr = lane & 15;
  const _Float16* arow0 = A + (size_t)(m0 + mr) * lda;
  const _Float16* arow1 = A + (size_t)(m0 + 16 + mr) * lda;
  const _Float16* brow  = Bt + (size_t)(n0 + mr) * ldb;

  v16h a0 = load_a16(arow0, lh);
  v16h a1 = load_a16(arow1, lh);
  v16h b[4];
#pragma unroll
  for (int t = 0; t < 4; ++t) b[t] = load_b16(brow + (size_t)(t * 16) * ldb, lh);

#pragma unroll 2
  for (int k0 = 0; k0 < K; k0 += 32) {
    int kn = (k0 + 32 < K) ? (k0 + 32) : 0;  // last iter: harmless reload
    v16h na0 = load_a16(arow0 + kn, lh);
    v16h na1 = load_a16(arow1 + kn, lh);
    v16h nb[4];
#pragma unroll
    for (int t = 0; t < 4; ++t)
      nb[t] = load_b16(brow + (size_t)(t * 16) * ldb + kn, lh);
#pragma unroll
    for (int t = 0; t < 4; ++t) {
      acc[0][t] = wmma_f16(a0, b[t], acc[0][t]);
      acc[1][t] = wmma_f16(a1, b[t], acc[1][t]);
    }
    a0 = na0;
    a1 = na1;
#pragma unroll
    for (int t = 0; t < 4; ++t) b[t] = nb[t];
  }
}

// ---------------------------------------------------------------------------
// 3) Q = LN(x) @ Wq + bq  -> Qh[b][h][sq][64] (f16)
// ---------------------------------------------------------------------------
__global__ void gemm_q_kernel(const _Float16* __restrict__ A,
                              const _Float16* __restrict__ Bt,
                              const float* __restrict__ bias,
                              _Float16* __restrict__ Qh) {
  int wid = (blockIdx.x * blockDim.x + threadIdx.x) >> 5;
  int lane = threadIdx.x & 31;
  const int NG = ND / 64;  // 16 n-groups
  int mt = wid / NG, ng = wid - mt * NG;
  int m0 = mt * 32, n0 = ng * 64;
  v8f acc[2][4] = {};
  gemm32x64(A, ND, Bt, ND, ND, m0, n0, lane, acc);
  int lh = lane >> 4, mr = lane & 15;
#pragma unroll
  for (int i = 0; i < 2; ++i)
#pragma unroll
    for (int t = 0; t < 4; ++t)
#pragma unroll
      for (int r = 0; r < 8; ++r) {
        int row = m0 + 16 * i + 8 * lh + r;
        int col = n0 + 16 * t + mr;
        int b = row >> 10, sq = row & 1023;
        int h = col >> 6, dq = col & 63;
        float v = acc[i][t][r] + bias[col];
        Qh[(((size_t)(b * NH + h) * NSQ) + sq) * NDH + dq] = (_Float16)v;
      }
}

// ---------------------------------------------------------------------------
// 4) KVp = LN(x) @ Wkv + bkv -> Kh[b][h][skv][64], Vt[b][h][64][skv] (f16)
// ---------------------------------------------------------------------------
__global__ void gemm_kv_kernel(const _Float16* __restrict__ A,
                               const _Float16* __restrict__ Bt,
                               const float* __restrict__ bias,
                               _Float16* __restrict__ Kh,
                               _Float16* __restrict__ Vt) {
  int wid = (blockIdx.x * blockDim.x + threadIdx.x) >> 5;
  int lane = threadIdx.x & 31;
  const int NG = 2048 / 64;  // 32 n-groups
  int mt = wid / NG, ng = wid - mt * NG;
  int m0 = mt * 32, n0 = ng * 64;
  v8f acc[2][4] = {};
  gemm32x64(A, ND, Bt, ND, ND, m0, n0, lane, acc);
  int lh = lane >> 4, mr = lane & 15;
#pragma unroll
  for (int i = 0; i < 2; ++i)
#pragma unroll
    for (int t = 0; t < 4; ++t)
#pragma unroll
      for (int r = 0; r < 8; ++r) {
        int row = m0 + 16 * i + 8 * lh + r;
        int col = n0 + 16 * t + mr;
        int b = row >> 10, sq = row & 1023;
        float v = acc[i][t][r] + bias[col];
        if (col < 1024) {
          int h = col >> 6, dq = col & 63;
          Kh[(((size_t)(b * NH + h) * NSQ) + sq) * NDH + dq] = (_Float16)v;
        } else {
          int nv = col - 1024;
          int h = nv >> 6, dv = nv & 63;
          Vt[(((size_t)(b * NH + h) * NDH) + dv) * NSQ + sq] = (_Float16)v;
        }
      }
}

// ---------------------------------------------------------------------------
// 5) Flash attention: one wave per (b, h, 16-query tile).
//    K/V chunks staged to double-buffered LDS with async global->LDS copies,
//    issued one chunk ahead.  Per 32-key chunk: 4 WMMAs (S=QK^T), fp32
//    online softmax, P staged via LDS into A-operand layout, 4 WMMAs (PV).
// ---------------------------------------------------------------------------
__global__ void attn_kernel(const _Float16* __restrict__ Qh,
                            const _Float16* __restrict__ Kh,
                            const _Float16* __restrict__ Vt,
                            const int* __restrict__ lens,
                            _Float16* __restrict__ Oh) {
  __shared__ __attribute__((aligned(16))) _Float16 kb[2][32 * 64];  // [key][dim]
  __shared__ __attribute__((aligned(16))) _Float16 vb[2][64 * 32];  // [dv][key]
  __shared__ __attribute__((aligned(16))) _Float16 pb[16 * 32];     // [qrow][key]
  int lane = threadIdx.x & 31;
  int lh = lane >> 4, mr = lane & 15;
  int tile = blockIdx.x;
  int qt = tile & 63;
  int bh = tile >> 6;   // b*NH + h
  int b = bh >> 4;
  int h = bh & 15;
  int q0 = qt * 16;
  int len = lens[b];
  int causal_end = q0 + 16;
  int kend = causal_end < len ? causal_end : len;
  int nchunks = (kend + 31) >> 5;

  const _Float16* kbb = Kh + (size_t)bh * NSQ * NDH;
  const _Float16* vbb = Vt + (size_t)bh * NDH * NSQ;

  // stage one 32-key chunk (K: 4KB contiguous, V: 64 rows x 64B) into buffer s
  auto stage = [&](int s, int kc) {
    const _Float16* ksrc = kbb + (size_t)kc * NDH;
#pragma unroll
    for (int i = 0; i < 8; ++i) {
      int f = i * 32 + lane;  // 0..255 16B chunks
      async_cp16(&kb[s][f * 8], ksrc + f * 8);
    }
#pragma unroll
    for (int i = 0; i < 8; ++i) {
      int f = i * 32 + lane;
      int row = f >> 2, part = f & 3;
      async_cp16(&vb[s][row * 32 + part * 8],
                 vbb + (size_t)row * NSQ + kc + part * 8);
    }
  };

  // Q A-operands for head dims 0..31 and 32..63 (loaded once)
  const _Float16* qrow = Qh + ((size_t)bh * NSQ + q0 + mr) * NDH;
  v16h aq0 = load_a16(qrow, lh);
  v16h aq1 = load_a16(qrow + 32, lh);

  float mrow[8], lrow[8];
  v8f O[4] = {};
#pragma unroll
  for (int r = 0; r < 8; ++r) { mrow[r] = -1e30f; lrow[r] = 0.f; }

  stage(0, 0);

  for (int ci = 0; ci < nchunks; ++ci) {
    int kc = ci * 32;
    int s = ci & 1;
    __syncthreads();  // prior reads of buffer s^1 done before restaging below
    if (ci + 1 < nchunks) {
      stage(s ^ 1, kc + 32);
      // async loads retire in order: with the next chunk's 16 transfers in
      // flight, <=16 outstanding means our chunk's 16 have landed.
      wait_async<16>();
    } else {
      wait_async<0>();
    }
    __syncthreads();

    // ---- S = (Q K^T) for two 16-key column tiles (B-operand from LDS) ----
    v8f S[2];
#pragma unroll
    for (int nt = 0; nt < 2; ++nt) {
      const _Float16* krow = &kb[s][(16 * nt + mr) * 64];
      v16h bk0 = load_b16(krow, lh);       // head dims 0..31
      v16h bk1 = load_b16(krow + 32, lh);  // head dims 32..63
      v8f sacc = {};
      sacc = wmma_f16(aq0, bk0, sacc);
      sacc = wmma_f16(aq1, bk1, sacc);
      S[nt] = sacc;
    }
    // ---- scale + causal/padding mask ----
    float sv[2][8];
#pragma unroll
    for (int nt = 0; nt < 2; ++nt) {
      int key = kc + 16 * nt + mr;
      bool kok = key < len;
#pragma unroll
      for (int r = 0; r < 8; ++r) {
        int qi = q0 + 8 * lh + r;
        float sval = S[nt][r] * 0.125f;  // 1/sqrt(64)
        sv[nt][r] = (kok && key <= qi) ? sval : -1e30f;
      }
    }
    // ---- online softmax (row = 8*lh + r, reduced over 16 lanes) ----
#pragma unroll
    for (int r = 0; r < 8; ++r) {
      float cm = fmaxf(sv[0][r], sv[1][r]);
      cm = fmaxf(cm, __shfl_xor(cm, 1, 32));
      cm = fmaxf(cm, __shfl_xor(cm, 2, 32));
      cm = fmaxf(cm, __shfl_xor(cm, 4, 32));
      cm = fmaxf(cm, __shfl_xor(cm, 8, 32));
      float mnew = fmaxf(mrow[r], cm);
      float alpha = __expf(mrow[r] - mnew);
      float p0 = __expf(sv[0][r] - mnew);
      float p1 = __expf(sv[1][r] - mnew);
      pb[(8 * lh + r) * 32 + mr]      = (_Float16)p0;
      pb[(8 * lh + r) * 32 + 16 + mr] = (_Float16)p1;
      float rs = p0 + p1;
      rs += __shfl_xor(rs, 1, 32);
      rs += __shfl_xor(rs, 2, 32);
      rs += __shfl_xor(rs, 4, 32);
      rs += __shfl_xor(rs, 8, 32);
      lrow[r] = lrow[r] * alpha + rs;
      mrow[r] = mnew;
#pragma unroll
      for (int t = 0; t < 4; ++t) O[t][r] *= alpha;
    }
    __syncthreads();  // P visible before cross-lane A-operand reads
    // ---- O += P @ V (K = 32 keys, 4 dv tiles, all operands from LDS) ----
    v16h ap = load_a16(&pb[mr * 32], lh);
#pragma unroll
    for (int t = 0; t < 4; ++t) {
      v16h bv = load_b16(&vb[s][(16 * t + mr) * 32], lh);
      O[t] = wmma_f16(ap, bv, O[t]);
    }
  }
  // ---- finalize: divide by row sums, store f16 [b*SQ+sq][h*64+dv] ----
#pragma unroll
  for (int r = 0; r < 8; ++r) {
    float inv = 1.0f / lrow[r];
    int row = b * NSQ + q0 + 8 * lh + r;
#pragma unroll
    for (int t = 0; t < 4; ++t) {
      Oh[(size_t)row * ND + h * NDH + 16 * t + mr] = (_Float16)(O[t][r] * inv);
    }
  }
}

// ---------------------------------------------------------------------------
// 6) out = attn @ Wo + bo + residual(LN(x))   (fp32 output)
// ---------------------------------------------------------------------------
__global__ void gemm_out_kernel(const _Float16* __restrict__ A,
                                const _Float16* __restrict__ Bt,
                                const float* __restrict__ bias,
                                const float* __restrict__ resid,
                                float* __restrict__ out) {
  int wid = (blockIdx.x * blockDim.x + threadIdx.x) >> 5;
  int lane = threadIdx.x & 31;
  const int NG = ND / 64;
  int mt = wid / NG, ng = wid - mt * NG;
  int m0 = mt * 32, n0 = ng * 64;
  v8f acc[2][4] = {};
  gemm32x64(A, ND, Bt, ND, ND, m0, n0, lane, acc);
  int lh = lane >> 4, mr = lane & 15;
#pragma unroll
  for (int i = 0; i < 2; ++i)
#pragma unroll
    for (int t = 0; t < 4; ++t)
#pragma unroll
      for (int r = 0; r < 8; ++r) {
        int row = m0 + 16 * i + 8 * lh + r;
        int col = n0 + 16 * t + mr;
        size_t idx = (size_t)row * ND + col;
        out[idx] = acc[i][t][r] + bias[col] + resid[idx];
      }
}

// ---------------------------------------------------------------------------
extern "C" void kernel_launch(void* const* d_in, const int* in_sizes, int n_in,
                              void* d_out, int out_size, void* d_ws,
                              size_t ws_size, hipStream_t stream) {
  (void)in_sizes; (void)n_in; (void)out_size; (void)ws_size;
  const float* x     = (const float*)d_in[0];
  // d_in[1] == x (self-attention; reference's equality check is true)
  const int*   lens  = (const int*)d_in[2];
  const float* Wq    = (const float*)d_in[3];
  const float* bq    = (const float*)d_in[4];
  const float* Wkv   = (const float*)d_in[5];
  const float* bkv   = (const float*)d_in[6];
  const float* Wo    = (const float*)d_in[7];
  const float* bo    = (const float*)d_in[8];
  const float* gamma = (const float*)d_in[9];
  const float* beta  = (const float*)d_in[10];
  float* out = (float*)d_out;

  char* ws = (char*)d_ws;
  float*    lnf   = (float*)(ws + 0);                       // 32 MB
  _Float16* lnh   = (_Float16*)(ws + 33554432);             // 16 MB
  _Float16* WqT   = (_Float16*)(ws + 50331648);             //  2 MB
  _Float16* WkvT  = (_Float16*)(ws + 52428800);             //  4 MB
  _Float16* WoT   = (_Float16*)(ws + 56623104);             //  2 MB
  _Float16* Qh    = (_Float16*)(ws + 58720256);             // 16 MB
  _Float16* Kh    = (_Float16*)(ws + 75497472);             // 16 MB
  _Float16* Vt    = (_Float16*)(ws + 92274688);             // 16 MB
  _Float16* attnh = (_Float16*)(ws + 109051904);            // 16 MB

  // 1) LayerNorm (8192 rows, wave per row)
  ln_kernel<<<NROWS / 8, 256, 0, stream>>>(x, gamma, beta, lnf, lnh);

  // 2) Weight transposes to f16
  wt_kernel<<<(ND * ND + 255) / 256, 256, 0, stream>>>(Wq, WqT, ND, ND);
  wt_kernel<<<(ND * 2048 + 255) / 256, 256, 0, stream>>>(Wkv, WkvT, ND, 2048);
  wt_kernel<<<(ND * ND + 255) / 256, 256, 0, stream>>>(Wo, WoT, ND, ND);

  // 3) Q projection: 256 m-tiles x 16 n-groups = 4096 waves
  gemm_q_kernel<<<512, 256, 0, stream>>>(lnh, WqT, bq, Qh);

  // 4) KV projection: 256 x 32 = 8192 waves
  gemm_kv_kernel<<<1024, 256, 0, stream>>>(lnh, WkvT, bkv, Kh, Vt);

  // 5) Attention: B*H*(SQ/16) = 8192 wave-sized blocks
  attn_kernel<<<NB * NH * (NSQ / 16), 32, 0, stream>>>(Qh, Kh, Vt, lens, attnh);

  // 6) Output projection + bias + residual
  gemm_out_kernel<<<512, 256, 0, stream>>>(attnh, WoT, bo, lnf, out);
}
